// ModifiedNAT_575525618129
// MI455X (gfx1250) — compile-verified, matrix operand order checked by reading, not run on hardware
//
#include <hip/hip_runtime.h>

// ---------------------------------------------------------------------------
// ModifiedNAT (2D neighborhood attention, K=7, NH=16, hd=32) for gfx1250.
//   1) cvt_kernel: f32 -> bf16 for q,k,v inputs and wq,wk,wv,wo weights (ws)
//   2) gemm_wmma<PLANAR>: X @ W^T + b  (v_wmma_f32_16x16x32_bf16),
//      async global->LDS double-buffered pipeline (GLOBAL_LOAD_ASYNC_TO_LDS),
//      writes planar bf16 [B, NH, hd, H, W]; q gets *hd^-0.5 in epilogue
//   3) natten2d: per-(head, 16x16 tile) attention, K/V neighborhood in LDS
//   4) gemm_wmma<ROWMAJOR_F32>: Xattn @ Wo^T + bo -> f32 d_out
// ---------------------------------------------------------------------------

using u16 = unsigned short;
using u32 = unsigned int;

typedef __attribute__((ext_vector_type(16))) __bf16 v16bf;
typedef __attribute__((ext_vector_type(8)))  float  v8f;
typedef __attribute__((ext_vector_type(4)))  int    v4i;

constexpr int MTOT = 32768;      // B*H*W
constexpr int CDIM = 512;        // C (= K = N of all GEMMs)
constexpr int HWI  = 16384;      // H*W per image
constexpr int HD   = 32;

#ifndef __has_builtin
#define __has_builtin(x) 0
#endif

#if __has_builtin(__builtin_amdgcn_global_load_async_to_lds_b128)
#define HAVE_ASYNC_LDS 1
#else
#define HAVE_ASYNC_LDS 0
#endif

__device__ __forceinline__ u16 f2bf(float f) {
    u32 u = __float_as_uint(f);
    u32 r = u + 0x7FFFu + ((u >> 16) & 1u);   // round-to-nearest-even
    return (u16)(r >> 16);
}
__device__ __forceinline__ float bf2f(u16 h) {
    return __uint_as_float(((u32)h) << 16);
}

#if HAVE_ASYNC_LDS
using gas_v4i = __attribute__((address_space(1))) v4i;   // global
using las_v4i = __attribute__((address_space(3))) v4i;   // LDS
__device__ __forceinline__ void async_cp16(u16* lds_dst, const u16* gsrc) {
    // GLOBAL_LOAD_ASYNC_TO_LDS_B128: LDS[dst] = MEM[src], tracked by ASYNCcnt
    __builtin_amdgcn_global_load_async_to_lds_b128(
        (gas_v4i*)gsrc, (las_v4i*)lds_dst, 0, 0);
}
#endif

template <int N>
__device__ __forceinline__ void wait_async() {
#if __has_builtin(__builtin_amdgcn_s_wait_asynccnt)
    __builtin_amdgcn_s_wait_asynccnt(N);
#else
    asm volatile("s_wait_asynccnt %0" ::"i"(N) : "memory");
#endif
}

// -------------------------------- convert ----------------------------------
__global__ __launch_bounds__(256) void cvt_kernel(const float* __restrict__ src,
                                                  u16* __restrict__ dst, int n4) {
    int i = blockIdx.x * blockDim.x + threadIdx.x;
    if (i < n4) {
        float4 f = ((const float4*)src)[i];
        uint2 o;
        o.x = (u32)f2bf(f.x) | ((u32)f2bf(f.y) << 16);
        o.y = (u32)f2bf(f.z) | ((u32)f2bf(f.w) << 16);
        ((uint2*)dst)[i] = o;
    }
}

// -------------------------------- GEMM -------------------------------------
// C[m,n] = sum_k A[m,k] * W[n,k]  (+ bias[n]); A:[MTOT,512] bf16, W:[512,512] bf16
// MODE 0: planar bf16 out, scaled:  out[b*C*HW + n*HW + hw] = bf16((acc+bias)*scale)
// MODE 1: row-major f32 out:        out[m*C + n] = acc + bias
template <int MODE>
__global__ __launch_bounds__(256) void gemm_wmma(const u16* __restrict__ A,
                                                 const u16* __restrict__ Bw,
                                                 const float* __restrict__ bias,
                                                 u16* __restrict__ outp,
                                                 float* __restrict__ outf,
                                                 float scale) {
#if HAVE_ASYNC_LDS
    __shared__ u16 As[2][128][48];   // 32 used + 16 pad halves -> 96B row stride
    __shared__ u16 Bs[2][128][48];
#else
    __shared__ u16 As[1][128][48];
    __shared__ u16 Bs[1][128][48];
#endif

    const int m0   = blockIdx.y * 128;
    const int n0   = blockIdx.x * 128;
    const int tid  = threadIdx.x;
    const int lane = tid & 31;
    const int wid  = tid >> 5;          // 8 waves
    const int wm   = (wid & 3) * 32;    // 4 waves along M
    const int wn   = (wid >> 2) * 64;   // 2 waves along N
    const int lr   = lane & 15;         // row/col within 16-tile
    const int kc   = (lane >> 4) * 16;  // k half-chunk per lane group

    v8f acc[2][4] = {};

#if HAVE_ASYNC_LDS
    // ---------------- async global->LDS double-buffered pipeline -----------
    auto issue_stage = [&](int buf, int k0) {
#pragma unroll
        for (int it = 0; it < 2; ++it) {
            int vv  = tid + it * 256;
            int row = vv >> 2;
            int ch  = (vv & 3) * 8;
            async_cp16(&As[buf][row][ch], A  + (size_t)(m0 + row) * CDIM + k0 + ch);
            async_cp16(&Bs[buf][row][ch], Bw + (size_t)(n0 + row) * CDIM + k0 + ch);
        }
    };

    issue_stage(0, 0);
    int buf = 0;
    for (int k0 = 0; k0 < CDIM; k0 += 32, buf ^= 1) {
        if (k0 + 32 < CDIM) {
            issue_stage(buf ^ 1, k0 + 32);
            wait_async<4>();   // the 4 older (current-stage) copies are done
        } else {
            wait_async<0>();
        }
        __syncthreads();

        v16bf af[2], bfv[4];
#pragma unroll
        for (int i = 0; i < 2; ++i) af[i]  = *(const v16bf*)&As[buf][wm + i * 16 + lr][kc];
#pragma unroll
        for (int j = 0; j < 4; ++j) bfv[j] = *(const v16bf*)&Bs[buf][wn + j * 16 + lr][kc];

#pragma unroll
        for (int i = 0; i < 2; ++i)
#pragma unroll
            for (int j = 0; j < 4; ++j)
                acc[i][j] = __builtin_amdgcn_wmma_f32_16x16x32_bf16(
                    false, af[i], false, bfv[j], (short)0, acc[i][j], false, false);
        __syncthreads();   // buf may be refilled two stages later
    }
#else
    // ---------------- synchronous fallback (VGPR round-trip) ---------------
    for (int k0 = 0; k0 < CDIM; k0 += 32) {
#pragma unroll
        for (int it = 0; it < 2; ++it) {
            int vv  = tid + it * 256;
            int row = vv >> 2;
            int ch  = (vv & 3) * 8;
            *(uint4*)&As[0][row][ch] = *(const uint4*)(A  + (size_t)(m0 + row) * CDIM + k0 + ch);
            *(uint4*)&Bs[0][row][ch] = *(const uint4*)(Bw + (size_t)(n0 + row) * CDIM + k0 + ch);
        }
        __syncthreads();

        v16bf af[2], bfv[4];
#pragma unroll
        for (int i = 0; i < 2; ++i) af[i]  = *(const v16bf*)&As[0][wm + i * 16 + lr][kc];
#pragma unroll
        for (int j = 0; j < 4; ++j) bfv[j] = *(const v16bf*)&Bs[0][wn + j * 16 + lr][kc];

#pragma unroll
        for (int i = 0; i < 2; ++i)
#pragma unroll
            for (int j = 0; j < 4; ++j)
                acc[i][j] = __builtin_amdgcn_wmma_f32_16x16x32_bf16(
                    false, af[i], false, bfv[j], (short)0, acc[i][j], false, false);
        __syncthreads();
    }
#endif

    // epilogue: per lane, fixed N = lr tile column; 8 consecutive M values
#pragma unroll
    for (int i = 0; i < 2; ++i) {
#pragma unroll
        for (int j = 0; j < 4; ++j) {
            int n  = n0 + wn + j * 16 + lr;
            int mb = m0 + wm + i * 16 + (lane >> 4) * 8;
            float bb = bias[n];
            if (MODE == 0) {
                u32 pk[4];
#pragma unroll
                for (int r = 0; r < 4; ++r) {
                    u16 lo = f2bf((acc[i][j][2 * r]     + bb) * scale);
                    u16 hi = f2bf((acc[i][j][2 * r + 1] + bb) * scale);
                    pk[r] = (u32)lo | ((u32)hi << 16);
                }
                int bi = mb >> 14;          // image index
                int hw = mb & (HWI - 1);    // pixel within image
                size_t addr = (size_t)bi * CDIM * HWI + (size_t)n * HWI + hw;
                *(uint4*)(outp + addr) = make_uint4(pk[0], pk[1], pk[2], pk[3]);
            } else {
                float* o = outf + (size_t)mb * CDIM + n;
#pragma unroll
                for (int r = 0; r < 8; ++r) o[(size_t)r * CDIM] = acc[i][j][r] + bb;
            }
        }
    }
}

// ------------------------------ attention ----------------------------------
// One block = one (b, nh) head and a 16x16 query tile. K/V neighborhood
// (<= 22x22x32 bf16 each) staged in LDS. Planar input layout:
//   addr = (bn*32 + c)*HWI + h*128 + w,  bn = b*16 + nh
__global__ __launch_bounds__(256) void natten2d(const u16* __restrict__ Qh,
                                                const u16* __restrict__ Kh,
                                                const u16* __restrict__ Vh,
                                                const float* __restrict__ rpb,
                                                u16* __restrict__ Xa) {
    __shared__ u16 Ks[22][22][32];
    __shared__ u16 Vs[22][22][32];

    const int bn = blockIdx.z;          // b*16 + nh
    const int nh = bn & 15;
    const int h0 = blockIdx.y * 16;
    const int w0 = blockIdx.x * 16;
    const size_t head = (size_t)bn * HD * HWI;

    const int rlo = max(h0 - 3, 0), rhi = min(h0 + 18, 127);
    const int clo = max(w0 - 3, 0), chi = min(w0 + 18, 127);
    const int rn = rhi - rlo + 1, cn = chi - clo + 1;
    const int tot = rn * cn;

    for (int i = threadIdx.x; i < HD * tot; i += 256) {
        int ch = i / tot;
        int rc = i - ch * tot;
        int r = rc / cn, c = rc - r * cn;
        size_t g = head + (size_t)ch * HWI + (size_t)(rlo + r) * 128 + (clo + c);
        Ks[r][c][ch] = Kh[g];
        Vs[r][c][ch] = Vh[g];
    }
    __syncthreads();

    const int ty = threadIdx.x >> 4, tx = threadIdx.x & 15;
    const int h = h0 + ty, w = w0 + tx;

    float qf[HD];
    {
        size_t qb = head + (size_t)h * 128 + w;
#pragma unroll
        for (int c = 0; c < HD; ++c) qf[c] = bf2f(Qh[qb + (size_t)c * HWI]);
    }

    const int sh = min(max(h - 3, 0), 121), sw = min(max(w - 3, 0), 121);
    const int ph = (h < 3) ? (6 - h) : ((h >= 125) ? (127 - h) : 3);
    const int pw = (w < 3) ? (6 - w) : ((w >= 125) ? (127 - w) : 3);

    float lg[49];
    float mx = -3.0e38f;
#pragma unroll
    for (int a = 0; a < 7; ++a) {
#pragma unroll
        for (int b2 = 0; b2 < 7; ++b2) {
            const u32* kp = (const u32*)&Ks[sh + a - rlo][sw + b2 - clo][0];
            float s = 0.f;
#pragma unroll
            for (int cc = 0; cc < 16; ++cc) {
                u32 pk = kp[cc];
                s += qf[2 * cc]     * bf2f((u16)(pk & 0xffffu));
                s += qf[2 * cc + 1] * bf2f((u16)(pk >> 16));
            }
            s += rpb[((size_t)nh * 13 + (ph + a)) * 13 + (pw + b2)];
            lg[a * 7 + b2] = s;
            mx = fmaxf(mx, s);
        }
    }

    float ssum = 0.f;
#pragma unroll
    for (int i = 0; i < 49; ++i) { lg[i] = __expf(lg[i] - mx); ssum += lg[i]; }
    const float inv = 1.f / ssum;

    float acc[HD];
#pragma unroll
    for (int c = 0; c < HD; ++c) acc[c] = 0.f;
#pragma unroll
    for (int a = 0; a < 7; ++a) {
#pragma unroll
        for (int b2 = 0; b2 < 7; ++b2) {
            float p = lg[a * 7 + b2] * inv;
            const u32* vp = (const u32*)&Vs[sh + a - rlo][sw + b2 - clo][0];
#pragma unroll
            for (int cc = 0; cc < 16; ++cc) {
                u32 pk = vp[cc];
                acc[2 * cc]     += p * bf2f((u16)(pk & 0xffffu));
                acc[2 * cc + 1] += p * bf2f((u16)(pk >> 16));
            }
        }
    }

    // write [M, C] row-major bf16: m = (b*128 + h)*128 + w, cols nh*32..nh*32+31
    const int b_ = bn >> 4;
    size_t m = ((size_t)b_ * 128 + h) * 128 + w;
    u16* op = Xa + m * CDIM + nh * HD;
    u32 pk[16];
#pragma unroll
    for (int cc = 0; cc < 16; ++cc)
        pk[cc] = (u32)f2bf(acc[2 * cc]) | ((u32)f2bf(acc[2 * cc + 1]) << 16);
    uint4* o4 = (uint4*)op;
    o4[0] = make_uint4(pk[0],  pk[1],  pk[2],  pk[3]);
    o4[1] = make_uint4(pk[4],  pk[5],  pk[6],  pk[7]);
    o4[2] = make_uint4(pk[8],  pk[9],  pk[10], pk[11]);
    o4[3] = make_uint4(pk[12], pk[13], pk[14], pk[15]);
}

// ------------------------------- launch ------------------------------------
extern "C" void kernel_launch(void* const* d_in, const int* in_sizes, int n_in,
                              void* d_out, int out_size, void* d_ws, size_t ws_size,
                              hipStream_t stream) {
    const float* q   = (const float*)d_in[0];
    const float* k   = (const float*)d_in[1];
    const float* v   = (const float*)d_in[2];
    const float* wq  = (const float*)d_in[3];
    const float* bq  = (const float*)d_in[4];
    const float* wk  = (const float*)d_in[5];
    const float* bk  = (const float*)d_in[6];
    const float* wv  = (const float*)d_in[7];
    const float* bvv = (const float*)d_in[8];
    const float* rpb = (const float*)d_in[9];
    const float* wo  = (const float*)d_in[10];
    const float* bo  = (const float*)d_in[11];

    char* ws = (char*)d_ws;
    const size_t MB = 1ull << 20;
    u16* Xq  = (u16*)(ws + 0 * MB);     // 32MB each
    u16* Xk  = (u16*)(ws + 32 * MB);
    u16* Xv  = (u16*)(ws + 64 * MB);
    u16* Qh  = (u16*)(ws + 96 * MB);    // planar projections
    u16* Khp = (u16*)(ws + 128 * MB);
    u16* Vhp = (u16*)(ws + 160 * MB);
    u16* Xa  = (u16*)(ws + 192 * MB);   // attention output [M, C]
    u16* Wq  = (u16*)(ws + 224 * MB);   // weights bf16, 512KB each
    u16* Wk  = Wq + (size_t)CDIM * CDIM;
    u16* Wv  = Wk + (size_t)CDIM * CDIM;
    u16* Wo  = Wv + (size_t)CDIM * CDIM;

    const int nX4 = MTOT * CDIM / 4;
    const int nW4 = CDIM * CDIM / 4;
    cvt_kernel<<<(nX4 + 255) / 256, 256, 0, stream>>>(q, Xq, nX4);
    cvt_kernel<<<(nX4 + 255) / 256, 256, 0, stream>>>(k, Xk, nX4);
    cvt_kernel<<<(nX4 + 255) / 256, 256, 0, stream>>>(v, Xv, nX4);
    cvt_kernel<<<(nW4 + 255) / 256, 256, 0, stream>>>(wq, Wq, nW4);
    cvt_kernel<<<(nW4 + 255) / 256, 256, 0, stream>>>(wk, Wk, nW4);
    cvt_kernel<<<(nW4 + 255) / 256, 256, 0, stream>>>(wv, Wv, nW4);
    cvt_kernel<<<(nW4 + 255) / 256, 256, 0, stream>>>(wo, Wo, nW4);

    const float scale = 0.17677669529663687f;  // 32^-0.5
    dim3 gg(CDIM / 128, MTOT / 128);
    gemm_wmma<0><<<gg, 256, 0, stream>>>(Xq, Wq, bq,  Qh,  nullptr, scale);
    gemm_wmma<0><<<gg, 256, 0, stream>>>(Xk, Wk, bk,  Khp, nullptr, 1.0f);
    gemm_wmma<0><<<gg, 256, 0, stream>>>(Xv, Wv, bvv, Vhp, nullptr, 1.0f);

    natten2d<<<dim3(8, 8, 32), 256, 0, stream>>>(Qh, Khp, Vhp, rpb, Xa);

    gemm_wmma<1><<<gg, 256, 0, stream>>>(Xa, Wo, bo, nullptr, (float*)d_out, 1.0f);
}
